// IntQwenHead_49615462204011
// MI455X (gfx1250) — 8-bit matrix operand order repaired and checked by reading, compiled, not run on hardware
//
#include <hip/hip_runtime.h>
#include <hip/hip_bf16.h>

typedef int v8i __attribute__((ext_vector_type(8)));

#define C_DIM   1024
#define VOCAB_N 151936
#define M_ROWS  64
#define KSTEP   64
#define NSTEPS  (C_DIM / KSTEP)
#define TRUNC_SH 9
#define OUT_SCALE 1.52587890625e-05f
#define ASLICE  (M_ROWS * C_DIM)

union FragI {
    v8i  v;
    int  i[8];
    long l[4];
};

__global__ void IntQwenHead_aquant(const float* __restrict__ x,
                                   signed char* __restrict__ adig) {
    int g = blockIdx.x * blockDim.x + threadIdx.x;
    if (g >= (M_ROWS * C_DIM / 8)) return;
    int row = g >> 7;
    int k0  = (g & 127) << 3;
    const float* xp = x + row * C_DIM + k0;
    float v[8];
    float amax = 0.f;
    #pragma unroll
    for (int i = 0; i < 8; ++i) { v[i] = xp[i]; amax = fmaxf(amax, fabsf(v[i])); }
    float t = amax * (1.0f / 127.0f);
    int e = 0;
    if (t > 1.0f) e = (int)ceilf(log2f(t));
    float inv = exp2f((float)(-e));
    #pragma unroll
    for (int i = 0; i < 8; ++i) {
        float q = rintf(v[i] * inv);
        q = fminf(127.f, fmaxf(-128.f, q));
        int xi = ((int)q) << e;
        int d0 = ((xi + 64) & 127) - 64; xi = (xi - d0) >> 7;
        int d1 = ((xi + 64) & 127) - 64; xi = (xi - d1) >> 7;
        int idx = row * C_DIM + k0 + i;
        adig[idx]              = (signed char)d0;
        adig[ASLICE + idx]     = (signed char)d1;
        adig[2 * ASLICE + idx] = (signed char)xi;
    }
}

__global__ __launch_bounds__(256, 1)
void IntQwenHead_gemm(const signed char* __restrict__ adig,
                      const int* __restrict__ wq,
                      const int* __restrict__ ws1,
                      const int* __restrict__ ws2,
                      float* __restrict__ out) {
    __shared__ int4 As4[768];

    const int tid   = threadIdx.x;
    const int lane  = tid & 31;
    const int wave  = tid >> 5;
    const int nbase = blockIdx.x * 128 + wave * 16;
    const int col   = nbase + (lane & 15);
    const int khalf = lane >> 4;
    const int arow  = lane & 15;
    const int ako   = khalf * 8;

    v8i acc[12];
    const v8i vzero = {0, 0, 0, 0, 0, 0, 0, 0};
    #pragma unroll
    for (int i = 0; i < 12; ++i) acc[i] = vzero;

    const long* Alds = (const long*)As4;

    for (int ks = 0; ks < NSTEPS; ++ks) {
        const int kb = ks * KSTEP;
        __syncthreads();
        {
            const int r   = tid >> 2;
            const int c16 = (tid & 3) << 4;
            #pragma unroll
            for (int s = 0; s < 3; ++s) {
                const int4* src = (const int4*)(adig + s * ASLICE + r * C_DIM + kb + c16);
                As4[s * 256 + tid] = *src;
            }
        }
        __syncthreads();

        FragI B;
        #pragma unroll
        for (int run = 0; run < 2; ++run) {
            const int krun = kb + khalf * 16 + run * 32;
            const int base = col * C_DIM + krun;
            int wv[16];
            const int4* wp = (const int4*)(wq + base);
            *(int4*)&wv[0]  = wp[0];
            *(int4*)&wv[4]  = wp[1];
            *(int4*)&wv[8]  = wp[2];
            *(int4*)&wv[12] = wp[3];
            const int gbase = base >> 3;
            int2 s1v = *(const int2*)(ws1 + gbase);
            int2 s2v = *(const int2*)(ws2 + gbase);
            const int m0 = (1 << s1v.x) + (1 << s2v.x);
            const int m1 = (1 << s1v.y) + (1 << s2v.y);
            #pragma unroll
            for (int d = 0; d < 4; ++d) {
                const int m = (d < 2) ? m0 : m1;
                int dw = 0;
                #pragma unroll
                for (int j = 0; j < 4; ++j) {
                    int p = wv[d * 4 + j] * m;
                    p = max(-128, min(127, p));
                    dw |= (p & 255) << (8 * j);
                }
                B.i[run * 4 + d] = dw;
            }
            if (ks + 1 < NSTEPS) __builtin_prefetch(wq + base + KSTEP, 0, 1);
        }

        #pragma unroll
        for (int mt = 0; mt < 4; ++mt) {
            #pragma unroll
            for (int s = 0; s < 3; ++s) {
                FragI A;
                const int b8 = s * 512 + (mt * 16 + arow) * 8 + (ako >> 2);
                A.l[0] = Alds[b8];
                A.l[1] = Alds[b8 + 1];
                A.l[2] = Alds[b8 + 4];
                A.l[3] = Alds[b8 + 5];
                acc[mt * 3 + s] = __builtin_amdgcn_wmma_i32_16x16x64_iu8(
                    true, A.v, true, B.v, acc[mt * 3 + s], false, false);
            }
        }
    }

    const int rsel = khalf * 8;
    #pragma unroll
    for (int mt = 0; mt < 4; ++mt) {
        #pragma unroll
        for (int j = 0; j < 8; ++j) {
            long y = (long)acc[mt * 3 + 0][j]
                   + ((long)acc[mt * 3 + 1][j] << 7)
                   + ((long)acc[mt * 3 + 2][j] << 14);
            y >>= TRUNC_SH;
            const int m = mt * 16 + j + rsel;
            out[(long)m * VOCAB_N + col] = (float)y * OUT_SCALE;
        }
    }
}

extern "C" void kernel_launch(void* const* d_in, const int* in_sizes, int n_in,
                              void* d_out, int out_size, void* d_ws, size_t ws_size,
                              hipStream_t stream) {
    const float* x   = (const float*)d_in[0];
    const int*   wq  = (const int*)d_in[1];
    const int*   ws1 = (const int*)d_in[2];
    const int*   ws2 = (const int*)d_in[3];
    float*       out = (float*)d_out;
    signed char* adig = (signed char*)d_ws;

    IntQwenHead_aquant<<<32, 256, 0, stream>>>(x, adig);
    IntQwenHead_gemm<<<VOCAB_N / 128, 256, 0, stream>>>(adig, wq, ws1, ws2, out);
}
